// TorchTuneLinearCrossEntropyLoss_19765439496671
// MI455X (gfx1250) — matrix operand order repaired and checked by reading, not compiled, not run on hardware
//
#include <hip/hip_runtime.h>
#include <hip/hip_bf16.h>
#include <math.h>

// ---------------------------------------------------------------------------
// Fused LM-head linear + cross-entropy (mean over valid tokens) for gfx1250.
//   logits = H @ W^T, loss = mean_n( logsumexp_v(logits[n,:]) - logits[n,t_n] )
// bf16 WMMA GEMM (f32 accumulate) with online logsumexp epilogue; exact fp32
// target logits; the [8192, 32000] logits matrix is never materialized.
// CDNA5 paths: v_wmma_f32_16x16x32_bf16, global_load_async_to_lds_b128 +
// s_wait_asynccnt for tile staging, global_prefetch for the weight stream.
// ---------------------------------------------------------------------------

typedef __attribute__((ext_vector_type(16))) __bf16         v16bf;
typedef __attribute__((ext_vector_type(8)))  float          v8f;
typedef __attribute__((ext_vector_type(8)))  unsigned short u16x8;
typedef __attribute__((ext_vector_type(4)))  unsigned short u16x4;

#define DM         2048            // d_model (reference constant)
#define LDS_STRIDE 2056            // DM + 8 bf16 pad -> 4112B row stride (no bank conflicts)
#define M_TILE     32              // token rows per workgroup

union FragBF {
    u16x8 h[2];
    v16bf v;
};

__device__ __forceinline__ unsigned short f2bf(float x) {
    unsigned u = __float_as_uint(x);
    u += 0x7FFFu + ((u >> 16) & 1u);          // round-to-nearest-even
    return (unsigned short)(u >> 16);
}

// --------------------------- fp32 -> bf16 conversion -----------------------
__global__ void cvt_bf16_kernel(const float4* __restrict__ in,
                                u16x4* __restrict__ out, int n4) {
    int i = blockIdx.x * blockDim.x + threadIdx.x;
    if (i < n4) {
        float4 v = in[i];
        u16x4 o;
        o.x = f2bf(v.x); o.y = f2bf(v.y); o.z = f2bf(v.z); o.w = f2bf(v.w);
        out[i] = o;
    }
}

// ----------------------- exact fp32 target logits --------------------------
// one wave (32 lanes) per token row
__global__ void tgt_logit_kernel(const float* __restrict__ hidden,
                                 const float* __restrict__ weight,
                                 const long long* __restrict__ tgt,
                                 float* __restrict__ out, int N, int D) {
    int row  = blockIdx.x * (blockDim.x >> 5) + (threadIdx.x >> 5);
    int lane = threadIdx.x & 31;
    if (row >= N) return;
    long long t  = tgt[row];
    long long tv = (t == -100) ? 0 : t;       // IGNORE_INDEX -> safe row 0
    const float4* h4 = (const float4*)(hidden + (size_t)row * D);
    const float4* w4 = (const float4*)(weight + (size_t)tv  * D);
    int n4 = D >> 2;
    float s = 0.f;
    for (int k = lane; k < n4; k += 32) {
        float4 a = h4[k], b = w4[k];
        s += a.x * b.x + a.y * b.y + a.z * b.z + a.w * b.w;
    }
#pragma unroll
    for (int m = 16; m >= 1; m >>= 1) s += __shfl_xor(s, m, 32);
    if (lane == 0) out[row] = s;
}

// --------------- WMMA bf16 GEMM + fused online logsumexp -------------------
// grid.x = N / M_TILE workgroups, 256 threads = 8 waves.
// Wave w sweeps vocab column-pairs [w*32, w*32+32) strided by 256 (V%256==0).
__global__ void lse_gemm_kernel(const unsigned short* __restrict__ hB,
                                const unsigned short* __restrict__ wB,
                                float* __restrict__ lse, int V) {
    extern __shared__ unsigned short smem[];   // M_TILE x LDS_STRIDE bf16
    const int tid     = threadIdx.x;
    const int lane    = tid & 31;
    const int wid     = tid >> 5;
    const int rowBase = blockIdx.x * M_TILE;

    // ---- stage 32 x 2048 bf16 hidden tile into padded LDS (async path) ----
    {
        const unsigned short* src = hB + (size_t)rowBase * DM;
        for (int c = tid; c < M_TILE * (DM / 8); c += blockDim.x) {
            int r  = c / (DM / 8);
            int k8 = c % (DM / 8);
            // LDS byte address = low 32 bits of the generic shared pointer
            unsigned lds_addr = (unsigned)(uintptr_t)(smem + r * LDS_STRIDE + k8 * 8);
            const unsigned short* g = src + (size_t)r * DM + (size_t)k8 * 8;
            asm volatile("global_load_async_to_lds_b128 %0, %1, off"
                         :: "v"(lds_addr), "v"(g) : "memory");
        }
        asm volatile("s_wait_asynccnt 0x0" ::: "memory");
    }
    __syncthreads();

    const int      cl = lane & 15;
    const bool     hi = lane >= 16;
    // A-fragment layout (16-bit A 16x32): lane<16 slots = K{kk..kk+7, kk+16..kk+23},
    // lane>=16 slots = K{kk+8..kk+15, kk+24..kk+31}
    const unsigned aK  = hi ? 8u : 0u;
    const unsigned aR0 = (unsigned)(cl)      * LDS_STRIDE;   // M-tile 0: rows 0..15
    const unsigned aR1 = (unsigned)(16 + cl) * LDS_STRIDE;   // M-tile 1: rows 16..31

    // per-lane online-softmax state: 16 rows (8 per M-tile, fixed by lane half)
    float mx[16], se[16];
#pragma unroll
    for (int i = 0; i < 16; ++i) { mx[i] = -__builtin_inff(); se[i] = 0.f; }

    for (int vb = wid * 32; vb < V; vb += 8 * 32) {
        // B-fragment layout (32x16): lane<16 = col cl, K{kk..kk+7, kk+16..kk+23}; lane>=16 = K{kk+8..kk+15, kk+24..kk+31}
        const unsigned short* bp0 = wB + ((size_t)(vb + cl)) * DM + (hi ? 8 : 0);
        const unsigned short* bp1 = bp0 + (size_t)16 * DM;
        // prefetch next iteration's weight rows (+256 vocab rows); speculative,
        // harmlessly dropped if past the end
        __builtin_prefetch(bp0 + (size_t)256 * DM, 0, 1);
        __builtin_prefetch(bp1 + (size_t)256 * DM, 0, 1);
        v8f acc00 = {}, acc01 = {}, acc10 = {}, acc11 = {};
#pragma unroll 4
        for (int kk = 0; kk < DM; kk += 32) {
            FragBF a0, a1, b0, b1;
            a0.h[0] = *(const u16x8*)(smem + aR0 + kk + aK);
            a0.h[1] = *(const u16x8*)(smem + aR0 + kk + 16 + aK);
            a1.h[0] = *(const u16x8*)(smem + aR1 + kk + aK);
            a1.h[1] = *(const u16x8*)(smem + aR1 + kk + 16 + aK);
            b0.h[0] = *(const u16x8*)(bp0 + kk);
            b0.h[1] = *(const u16x8*)(bp0 + kk + 16);
            b1.h[0] = *(const u16x8*)(bp1 + kk);
            b1.h[1] = *(const u16x8*)(bp1 + kk + 16);
            acc00 = __builtin_amdgcn_wmma_f32_16x16x32_bf16(false, a0.v, false, b0.v, (short)0, acc00, false, false);
            acc10 = __builtin_amdgcn_wmma_f32_16x16x32_bf16(false, a1.v, false, b0.v, (short)0, acc10, false, false);
            acc01 = __builtin_amdgcn_wmma_f32_16x16x32_bf16(false, a0.v, false, b1.v, (short)0, acc01, false, false);
            acc11 = __builtin_amdgcn_wmma_f32_16x16x32_bf16(false, a1.v, false, b1.v, (short)0, acc11, false, false);
        }
        // ---- online max/sum-exp update (C/D layout: VGPR r -> row r | r+8) ----
#pragma unroll
        for (int r = 0; r < 8; ++r) {
            float x, nm;
            x = acc00[r]; nm = fmaxf(mx[r], x);
            se[r] = se[r] * __expf(mx[r] - nm) + __expf(x - nm); mx[r] = nm;
            x = acc01[r]; nm = fmaxf(mx[r], x);
            se[r] = se[r] * __expf(mx[r] - nm) + __expf(x - nm); mx[r] = nm;
            int i = 8 + r;
            x = acc10[r]; nm = fmaxf(mx[i], x);
            se[i] = se[i] * __expf(mx[i] - nm) + __expf(x - nm); mx[i] = nm;
            x = acc11[r]; nm = fmaxf(mx[i], x);
            se[i] = se[i] * __expf(mx[i] - nm) + __expf(x - nm); mx[i] = nm;
        }
    }

    // ---- reduce over the 16 column lanes of each half-wave ----
#pragma unroll
    for (int i = 0; i < 16; ++i) {
#pragma unroll
        for (int m = 1; m <= 8; m <<= 1) {
            float om = __shfl_xor(mx[i], m, 32);
            float os = __shfl_xor(se[i], m, 32);
            float nm = fmaxf(mx[i], om);
            se[i] = se[i] * __expf(mx[i] - nm) + os * __expf(om - nm);
            mx[i] = nm;
        }
    }

    __syncthreads();                 // tile reads done -> reuse LDS as scratch
    float* scr = (float*)smem;       // [8 waves][32 rows][2]
    if (cl == 0) {
        int r0 = hi ? 8 : 0;
#pragma unroll
        for (int i = 0; i < 16; ++i) {
            int row = (i < 8) ? (r0 + i) : (16 + r0 + (i - 8));
            scr[(wid * 32 + row) * 2 + 0] = mx[i];
            scr[(wid * 32 + row) * 2 + 1] = se[i];
        }
    }
    __syncthreads();
    if (tid < 32) {                  // combine 8 waves, emit logsumexp
        float m = -__builtin_inff(), s = 0.f;
#pragma unroll
        for (int w = 0; w < 8; ++w) {
            float om = scr[(w * 32 + tid) * 2 + 0];
            float os = scr[(w * 32 + tid) * 2 + 1];
            float nm = fmaxf(m, om);
            s = s * __expf(m - nm) + os * __expf(om - nm);
            m = nm;
        }
        lse[rowBase + tid] = m + __logf(s);
    }
}

// ------------------------------ final reduce -------------------------------
__global__ void finalize_kernel(const float* __restrict__ lse,
                                const float* __restrict__ tgl,
                                const long long* __restrict__ tgt,
                                float* __restrict__ out, int N) {
    __shared__ float ssum[256];
    __shared__ int   scnt[256];
    int tid = threadIdx.x;
    float s = 0.f; int c = 0;
    for (int i = tid; i < N; i += blockDim.x) {
        long long t = tgt[i];
        if (t != -100) { s += lse[i] - tgl[i]; ++c; }
    }
    ssum[tid] = s; scnt[tid] = c;
    __syncthreads();
    for (int off = 128; off > 0; off >>= 1) {
        if (tid < off) { ssum[tid] += ssum[tid + off]; scnt[tid] += scnt[tid + off]; }
        __syncthreads();
    }
    if (tid == 0) out[0] = (scnt[0] > 0) ? (ssum[0] / (float)scnt[0]) : 0.f;
}

// ---------------------------------------------------------------------------
extern "C" void kernel_launch(void* const* d_in, const int* in_sizes, int n_in,
                              void* d_out, int out_size, void* d_ws, size_t ws_size,
                              hipStream_t stream) {
    (void)n_in; (void)out_size; (void)ws_size;
    const float*     hidden  = (const float*)d_in[0];      // [N, D] fp32
    const long long* targets = (const long long*)d_in[1];  // [N] int64
    const float*     weight  = (const float*)d_in[2];      // [V, D] fp32
    const int N = in_sizes[1];             // 8192 tokens
    const int D = in_sizes[0] / N;         // 2048
    const int V = in_sizes[2] / D;         // 32000

    // workspace layout
    char* ws = (char*)d_ws;
    unsigned short* wB = (unsigned short*)ws;               // V*D bf16
    size_t off = (size_t)V * D * 2;
    unsigned short* hB = (unsigned short*)(ws + off);       // N*D bf16
    off += (size_t)N * D * 2;
    float* lse = (float*)(ws + off); off += (size_t)N * 4;
    float* tgl = (float*)(ws + off); off += (size_t)N * 4;

    // 1) fp32 -> bf16 (RNE) for weight and hidden
    {
        int n4 = (V * D) / 4;
        cvt_bf16_kernel<<<(n4 + 255) / 256, 256, 0, stream>>>((const float4*)weight, (u16x4*)wB, n4);
        int m4 = (N * D) / 4;
        cvt_bf16_kernel<<<(m4 + 255) / 256, 256, 0, stream>>>((const float4*)hidden, (u16x4*)hB, m4);
    }
    // 2) exact fp32 target logits (one wave per row)
    tgt_logit_kernel<<<(N + 7) / 8, 256, 0, stream>>>(hidden, weight, targets, tgl, N, D);
    // 3) WMMA bf16 GEMM + fused online logsumexp
    {
        size_t lds_bytes = (size_t)M_TILE * LDS_STRIDE * 2;   // 131,584 B (< 320 KB WGP LDS)
        (void)hipFuncSetAttribute((const void*)lse_gemm_kernel,
                                  hipFuncAttributeMaxDynamicSharedMemorySize,
                                  (int)lds_bytes);
        lse_gemm_kernel<<<N / M_TILE, 256, lds_bytes, stream>>>(hB, wB, lse, V);
    }
    // 4) mean over valid tokens
    finalize_kernel<<<1, 256, 0, stream>>>(lse, tgl, targets, (float*)d_out, N);
}
